// DenceGCN_14310831030370
// MI455X (gfx1250) — matrix operand order repaired and checked by reading, not compiled, not run on hardware
//
#include <hip/hip_runtime.h>
#include <math.h>

typedef __attribute__((ext_vector_type(16))) __bf16 v16bf;
typedef __attribute__((ext_vector_type(8)))  float  v8f;

#define NNODES 100000
#define FEAT   512
#define HID    64
#define NCLS   40
#define NEDGE  1600000
#define NLAYER 6
#define LDH    896        // hcat row stride = 512 + 6*64

static __device__ __forceinline__ void atomAddF(float* p, float v) {
  __hip_atomic_fetch_add(p, v, __ATOMIC_RELAXED, __HIP_MEMORY_SCOPE_AGENT);
}

// ---------------- utility kernels ----------------
__global__ void k_fill(float* p, float v, int n) {
  int i = blockIdx.x * blockDim.x + threadIdx.x;
  if (i < n) p[i] = v;
}

__global__ void k_copy_x(const float* __restrict__ x, float* __restrict__ hcat) {
  int i = blockIdx.x * blockDim.x + threadIdx.x;
  if (i < NNODES * FEAT) {
    int n = i / FEAT, j = i - n * FEAT;
    hcat[(size_t)n * LDH + j] = x[i];
  }
}

__global__ void k_deg(const int* __restrict__ ei, float* __restrict__ deg) {
  int e = blockIdx.x * blockDim.x + threadIdx.x;
  if (e < NEDGE) atomAddF(&deg[ei[NEDGE + e]], 1.0f);   // col = target
}

__global__ void k_rsqrt_inplace(float* d, int n) {
  int i = blockIdx.x * blockDim.x + threadIdx.x;
  if (i < n) d[i] = rsqrtf(d[i]);                       // deg >= 1 always (self loop)
}

// Pack W [K, ncolsSrc] fp32 into per-lane WMMA B-fragment bf16 layout.
// Fragment (ks,nt): element (lane,i) = W[ks*32 + (i>>3)*16 + (lane>>4)*8 + (i&7)][nt*16 + (lane&15)]
__global__ void k_pack_w(const float* __restrict__ W, __bf16* __restrict__ Bp,
                         int ncolsSrc, int NT, int total) {
  int t = blockIdx.x * blockDim.x + threadIdx.x;
  if (t >= total) return;
  int i    = t & 15;
  int lane = (t >> 4) & 31;
  int frag = t >> 9;
  int ks = frag / NT, nt = frag - ks * NT;
  int kk = ks * 32 + ((i >> 3) << 4) + ((lane >> 4) << 3) + (i & 7);
  int nn = nt * 16 + (lane & 15);
  float v = (nn < ncolsSrc) ? W[(size_t)kk * ncolsSrc + nn] : 0.0f;
  Bp[t] = (__bf16)v;
}

// ---------------- WMMA GEMM: D[M,ncolsOut] = A[M,K](fp32, row stride lda) * Bpacked ----------------
// One wave computes a 16 x (NT*16) tile. K multiple of 32. M = NNODES (multiple of 16).
template <int NT>
__global__ void k_gemm_bf16(const float* __restrict__ A, int lda,
                            const __bf16* __restrict__ Bp,
                            float* __restrict__ D, int ldd,
                            int K, int ncolsOut) {
  int wave = blockIdx.x * (blockDim.x >> 5) + (threadIdx.x >> 5);
  if (wave * 16 >= NNODES) return;            // wave-uniform: EXEC stays all-1 for WMMA
  int lane = threadIdx.x & 31;
  int m    = wave * 16 + (lane & 15);
  int h    = lane >> 4;

  v8f acc[NT];
#pragma unroll
  for (int t = 0; t < NT; ++t) acc[t] = v8f{0.f, 0.f, 0.f, 0.f, 0.f, 0.f, 0.f, 0.f};

  const float* Arow = A + (size_t)m * lda;
  int ksteps = K >> 5;
  for (int ks = 0; ks < ksteps; ++ks) {
    int c0 = ks * 32 + h * 8;
    float4 f0 = *(const float4*)(Arow + c0);
    float4 f1 = *(const float4*)(Arow + c0 + 4);
    float4 f2 = *(const float4*)(Arow + c0 + 16);
    float4 f3 = *(const float4*)(Arow + c0 + 20);
    v16bf a;
    a[0] = (__bf16)f0.x; a[1] = (__bf16)f0.y; a[2]  = (__bf16)f0.z; a[3]  = (__bf16)f0.w;
    a[4] = (__bf16)f1.x; a[5] = (__bf16)f1.y; a[6]  = (__bf16)f1.z; a[7]  = (__bf16)f1.w;
    a[8] = (__bf16)f2.x; a[9] = (__bf16)f2.y; a[10] = (__bf16)f2.z; a[11] = (__bf16)f2.w;
    a[12] = (__bf16)f3.x; a[13] = (__bf16)f3.y; a[14] = (__bf16)f3.z; a[15] = (__bf16)f3.w;

    const __bf16* bbase = Bp + ((size_t)ks * NT) * 512 + lane * 16;
#pragma unroll
    for (int nt = 0; nt < NT; ++nt) {
      v16bf b = *(const v16bf*)(bbase + nt * 512);
      acc[nt] = __builtin_amdgcn_wmma_f32_16x16x32_bf16(
          /*neg_a=*/false, a, /*neg_b=*/false, b,
          /*c_mod=*/(short)0, acc[nt], /*reuse_a=*/false, /*reuse_b=*/false);
    }
  }

  int n0 = lane & 15;
#pragma unroll
  for (int nt = 0; nt < NT; ++nt) {
    int col = nt * 16 + n0;
    if (col < ncolsOut) {
#pragma unroll
      for (int v = 0; v < 8; ++v) {
        int row = wave * 16 + v + 8 * h;
        D[(size_t)row * ldd + col] = acc[nt][v];
      }
    }
  }
}

// ---------------- edge aggregation: agg[col] += lin[row] * dinv[row]*dinv[col] ----------------
// One wave per edge (E real edges + N self loops). Lane t handles features 2t, 2t+1.
__global__ void k_aggregate(const int* __restrict__ ei, const float* __restrict__ dinv,
                            const float* __restrict__ lin, float* __restrict__ agg) {
  int wid = blockIdx.x * (blockDim.x >> 5) + (threadIdx.x >> 5);
  if (wid >= NEDGE + NNODES) return;
  int t = threadIdx.x & 31;
  int r, c;
  float nrm;
  if (wid < NEDGE) {
    r = ei[wid];
    c = ei[NEDGE + wid];
    nrm = dinv[r] * dinv[c];
  } else {
    r = c = wid - NEDGE;
    float d = dinv[r];
    nrm = d * d;
  }
  float2 v = *(const float2*)(lin + (size_t)r * HID + t * 2);
  float* dst = agg + (size_t)c * HID + t * 2;
  atomAddF(dst, v.x * nrm);
  atomAddF(dst + 1, v.y * nrm);
}

__global__ void k_bias_relu(const float* __restrict__ agg, const float* __restrict__ b,
                            float* __restrict__ hcat, int colOff) {
  int i = blockIdx.x * blockDim.x + threadIdx.x;
  if (i >= NNODES * HID) return;
  int n = i >> 6, j = i & 63;
  float v = agg[i] + b[j];
  hcat[(size_t)n * LDH + colOff + j] = v > 0.f ? v : 0.f;
}

// ---------------- row-wise log-softmax (wave per row, C=40) ----------------
__global__ void k_log_softmax(const float* __restrict__ logits, const float* __restrict__ blin,
                              float* __restrict__ out) {
  int row = blockIdx.x * (blockDim.x >> 5) + (threadIdx.x >> 5);
  if (row >= NNODES) return;
  int j = threadIdx.x & 31;
  float l0 = (j < NCLS) ? logits[(size_t)row * NCLS + j] + blin[j] : -INFINITY;
  float l1 = (j + 32 < NCLS) ? logits[(size_t)row * NCLS + j + 32] + blin[j + 32] : -INFINITY;
  float m = fmaxf(l0, l1);
#pragma unroll
  for (int o = 16; o >= 1; o >>= 1) m = fmaxf(m, __shfl_xor(m, o, 32));
  float s = expf(l0 - m) + expf(l1 - m);
#pragma unroll
  for (int o = 16; o >= 1; o >>= 1) s += __shfl_xor(s, o, 32);
  float lse = m + logf(s);
  if (j < NCLS) out[(size_t)row * NCLS + j] = l0 - lse;
  if (j + 32 < NCLS) out[(size_t)row * NCLS + j + 32] = l1 - lse;
}

// ---------------- host launcher ----------------
extern "C" void kernel_launch(void* const* d_in, const int* in_sizes, int n_in,
                              void* d_out, int out_size, void* d_ws, size_t ws_size,
                              hipStream_t stream) {
  const float* x  = (const float*)d_in[0];
  const int*   ei = (const int*)d_in[1];
  const float* W[NLAYER];
  const float* bb[NLAYER];
  for (int i = 0; i < NLAYER; ++i) {
    W[i]  = (const float*)d_in[2 + 2 * i];
    bb[i] = (const float*)d_in[3 + 2 * i];
  }
  const float* Wlin = (const float*)d_in[14];
  const float* blin = (const float*)d_in[15];
  float* out = (float*)d_out;

  char* ws = (char*)d_ws;
  size_t off = 0;
  auto alloc = [&](size_t bytes) -> void* {
    void* p = ws + off;
    off = (off + bytes + 255) & ~(size_t)255;
    return p;
  };
  float*  hcat = (float*)alloc((size_t)NNODES * LDH * 4);   // 358.4 MB
  float*  lin  = (float*)alloc((size_t)NNODES * HID * 4);   // 25.6 MB (reused for logits)
  float*  agg  = (float*)alloc((size_t)NNODES * HID * 4);   // 25.6 MB
  float*  dinv = (float*)alloc((size_t)NNODES * 4);
  __bf16* Bp   = (__bf16*)alloc(131072 * 2);                // packed W fragments
  float*  logits = lin;                                     // reuse after last layer

  const int TPB = 256;
  // degree -> dinv (self loops contribute 1)
  k_fill<<<(NNODES + TPB - 1) / TPB, TPB, 0, stream>>>(dinv, 1.0f, NNODES);
  k_deg<<<(NEDGE + TPB - 1) / TPB, TPB, 0, stream>>>(ei, dinv);
  k_rsqrt_inplace<<<(NNODES + TPB - 1) / TPB, TPB, 0, stream>>>(dinv, NNODES);
  // hcat[:, 0:512] = x
  k_copy_x<<<(NNODES * FEAT + TPB - 1) / TPB, TPB, 0, stream>>>(x, hcat);

  const int mtiles = NNODES / 16;                 // 6250, exact
  const int gemmBlocks = (mtiles + 3) / 4;        // 4 waves (128 threads) per block
  const int edgeWaves = NEDGE + NNODES;
  const int aggBlocks = (edgeWaves + 7) / 8;      // 8 waves per 256-thread block

  for (int i = 0; i < NLAYER; ++i) {
    int K = FEAT + HID * i;                       // 512..832, multiple of 32
    int total = (K / 32) * 4 * 512;
    k_pack_w<<<(total + TPB - 1) / TPB, TPB, 0, stream>>>(W[i], Bp, HID, 4, total);
    k_gemm_bf16<4><<<gemmBlocks, 128, 0, stream>>>(hcat, LDH, Bp, lin, HID, K, HID);
    k_fill<<<(NNODES * HID + TPB - 1) / TPB, TPB, 0, stream>>>(agg, 0.0f, NNODES * HID);
    k_aggregate<<<aggBlocks, 256, 0, stream>>>(ei, dinv, lin, agg);
    k_bias_relu<<<(NNODES * HID + TPB - 1) / TPB, TPB, 0, stream>>>(agg, bb[i], hcat, K);
  }

  // JK head: hcat[N,896] @ Wlin[896,40] (padded to 48 cols), then log-softmax
  {
    int K = LDH;                                  // 896
    int total = (K / 32) * 3 * 512;
    k_pack_w<<<(total + TPB - 1) / TPB, TPB, 0, stream>>>(Wlin, Bp, NCLS, 3, total);
    k_gemm_bf16<3><<<gemmBlocks, 128, 0, stream>>>(hcat, LDH, Bp, logits, NCLS, K, NCLS);
    int lsmBlocks = (NNODES + 7) / 8;             // 8 rows per 256-thread block
    k_log_softmax<<<lsmBlocks, 256, 0, stream>>>(logits, blin, out);
  }
}
